// AutoDiscretizationEmbedding_62749472195319
// MI455X (gfx1250) — matrix operand order repaired and checked by reading, not compiled, not run on hardware
//
#include <hip/hip_runtime.h>
#include <hip/hip_bf16.h>

typedef __attribute__((ext_vector_type(16))) _Float16 v16h;
typedef __attribute__((ext_vector_type(8)))  float    v8f;

#define BINS     100
#define KPAD     128      // padded K (bins) for WMMA
#define DIM      512
#define TM       32       // tokens per block
#define NTHREADS 256      // 8 wave32s
#define LDK      132      // padded LDS row stride (halves) to dodge bank conflicts
#define SLABS    4        // DIM / SLABW
#define SLABW    128

union Frag { v16h v; unsigned int u[8]; };

// A fragment: 16x32 f16. Lane m(0-15) holds K=0..15, lanes 16-31 hold K=16..31
// (VGPR r = K pair {2r,2r+1}). hA rows are K-contiguous.
__device__ __forceinline__ v16h load_a_frag(const _Float16 (*hA)[LDK], int rt, int kc, int lane) {
    Frag a;
    const int m    = lane & 15;
    const int koff = (lane >> 4) ? 16 : 0;
    const _Float16* row = &hA[rt * 16 + m][kc * 32 + koff];
#pragma unroll
    for (int r = 0; r < 8; ++r) {
        const int k = 2 * r;
        a.u[r] = *(const unsigned int*)(row + k);
    }
    return a.v;
}

// B fragment: 32x16 f16. Lane n(0-15) holds K=0..15, lanes 16-31 hold K=16..31.
// Bst is staged transposed (Bst[n][k], k contiguous) so each lane reads 32 contiguous bytes.
__device__ __forceinline__ v16h load_b_frag(const _Float16 (*Bst)[LDK], int nt, int kc, int lane) {
    Frag b;
    const int n    = lane & 15;
    const int koff = (lane >> 4) ? 16 : 0;
    const _Float16* col = &Bst[nt * 16 + n][kc * 32 + koff];
#pragma unroll
    for (int r = 0; r < 8; ++r) b.u[r] = *(const unsigned int*)(col + 2 * r);
    return b.v;
}

__global__ __launch_bounds__(NTHREADS)
void auto_disc_emb_kernel(const float* __restrict__ x,
                          const float* __restrict__ w1,
                          const float* __restrict__ b1,
                          const float* __restrict__ w2,
                          const float* __restrict__ b2,
                          const float* __restrict__ emb,
                          float* __restrict__ out,
                          int ntok) {
    __shared__ _Float16 hA[TM][LDK];      // h (f16) -> later softmax weights (A operand)
    __shared__ _Float16 Bst[KPAD][LDK];   // B staging: w2^T, then emb slabs (k-contiguous)
    __shared__ float    logits[TM][KPAD];
    __shared__ float    x_s[TM];
    __shared__ float    w1_s[KPAD], b1_s[KPAD], b2_s[KPAD];
    __shared__ float    redmax[TM][8], redsum[TM][8];
    __shared__ float    rowmax[TM], rowsum[TM];

    const int tid   = threadIdx.x;
    const int lane  = tid & 31;
    const int wv    = tid >> 5;          // 0..7
    const int tbase = blockIdx.x * TM;

    // ---------- Phase 0: constants + x tile ----------
    if (tid < KPAD) {
        w1_s[tid] = (tid < BINS) ? w1[tid] : 0.0f;
        b1_s[tid] = (tid < BINS) ? b1[tid] : 0.0f;
        b2_s[tid] = (tid < BINS) ? b2[tid] : 0.0f;
    }
    if (tid < TM) {
        const int t = (tbase + tid < ntok) ? (tbase + tid) : (ntok - 1);
        x_s[tid] = x[t];
    }
    __syncthreads();

    // ---------- Phase 1: h = leakyrelu(x*w1+b1) -> f16 A buffer; stage w2^T -> f16 B ----------
    for (int idx = tid; idx < TM * KPAD; idx += NTHREADS) {
        const int t = idx >> 7, j = idx & (KPAD - 1);
        float v = 0.0f;
        if (j < BINS) {
            v = fmaf(x_s[t], w1_s[j], b1_s[j]);
            v = (v >= 0.0f) ? v : 0.1f * v;          // LeakyReLU(0.1)
        }
        hA[t][j] = (_Float16)v;
    }
    for (int idx = tid; idx < KPAD * KPAD; idx += NTHREADS) {
        const int j = idx >> 7, k = idx & (KPAD - 1);          // B[k][j] = w2[j][k]
        const float v = (j < BINS && k < BINS) ? w2[j * BINS + k] : 0.0f;
        Bst[j][k] = (_Float16)v;
    }
    __syncthreads();

    // ---------- Phase 2: cross = h @ w2^T via WMMA; logits = h + cross + b2 ----------
    {
        const int rt  = wv & 1;              // 2 row tiles of 16 tokens
        const int ct0 = (wv >> 1) * 2;       // 8 col tiles over 4 wave groups
        for (int c = 0; c < 2; ++c) {
            const int ct = ct0 + c;
            v8f acc = {};
#pragma unroll
            for (int kc = 0; kc < 4; ++kc) {
                v16h a = load_a_frag(hA, rt, kc, lane);
                v16h b = load_b_frag(Bst, ct, kc, lane);
                acc = __builtin_amdgcn_wmma_f32_16x16x32_f16(
                        false, a, false, b, (short)0, acc, false, false);
            }
            // C layout: VGPR r, lanes 0-15 -> M=r ; lanes 16-31 -> M=r+8 ; N = lane&15
            const int n2 = lane & 15;
            const int mh = (lane >> 4) * 8;
#pragma unroll
            for (int r = 0; r < 8; ++r) {
                const int t = rt * 16 + mh + r;
                const int j = ct * 16 + n2;
                float lg;
                if (j < BINS) {
                    float hv = fmaf(x_s[t], w1_s[j], b1_s[j]);
                    hv = (hv >= 0.0f) ? hv : 0.1f * hv;
                    lg = acc[r] + b2_s[j] + hv;      // BIN_ALPHA = 1.0 residual
                } else {
                    lg = -3.0e38f;                   // padded bins -> softmax weight 0
                }
                logits[t][j] = lg;
            }
        }
    }
    __syncthreads();

    // ---------- Phase 3: softmax over bins; weights back into f16 A buffer ----------
    {
        const int t = tid >> 3;              // 8 threads per token
        const int q = tid & 7;               // each covers 16 padded bins
        float mx = -3.0e38f;
        for (int j = q * 16; j < q * 16 + 16; ++j) mx = fmaxf(mx, logits[t][j]);
        redmax[t][q] = mx;
        __syncthreads();
        if (q == 0) {
            float m2 = redmax[t][0];
            for (int i = 1; i < 8; ++i) m2 = fmaxf(m2, redmax[t][i]);
            rowmax[t] = m2;
        }
        __syncthreads();
        const float rm = rowmax[t];
        float s = 0.0f;
        for (int j = q * 16; j < q * 16 + 16; ++j) {
            const float e = __expf(logits[t][j] - rm);
            logits[t][j] = e;
            s += e;
        }
        redsum[t][q] = s;
        __syncthreads();
        if (q == 0) {
            float s2 = 0.0f;
            for (int i = 0; i < 8; ++i) s2 += redsum[t][i];
            rowsum[t] = s2;
        }
        __syncthreads();
        const float inv = 1.0f / rowsum[t];
        for (int j = q * 16; j < q * 16 + 16; ++j)
            hA[t][j] = (_Float16)(logits[t][j] * inv);
    }
    __syncthreads();

    // ---------- Phase 4: out = weight @ emb via WMMA, 4 slabs of 128 dims ----------
    for (int slab = 0; slab < SLABS; ++slab) {
        // stage emb slab transposed: Bst[d][k] = emb[k][slab*128 + d] (global reads coalesced)
        for (int idx = tid; idx < KPAD * SLABW; idx += NTHREADS) {
            const int k = idx >> 7, d = idx & (SLABW - 1);
            const float v = (k < BINS) ? emb[k * DIM + slab * SLABW + d] : 0.0f;
            Bst[d][k] = (_Float16)v;
        }
        // warm next slab into cache while this one computes
        if (slab + 1 < SLABS && tid < BINS)
            __builtin_prefetch(&emb[tid * DIM + (slab + 1) * SLABW], 0, 1);
        __syncthreads();

        const int rt  = wv & 1;
        const int ct0 = (wv >> 1) * 2;
        for (int c = 0; c < 2; ++c) {
            const int ct = ct0 + c;
            v8f acc = {};
#pragma unroll
            for (int kc = 0; kc < 4; ++kc) {
                v16h a = load_a_frag(hA, rt, kc, lane);
                v16h b = load_b_frag(Bst, ct, kc, lane);
                acc = __builtin_amdgcn_wmma_f32_16x16x32_f16(
                        false, a, false, b, (short)0, acc, false, false);
            }
            const int n2 = lane & 15;
            const int mh = (lane >> 4) * 8;
#pragma unroll
            for (int r = 0; r < 8; ++r) {
                const int t = tbase + rt * 16 + mh + r;
                const int d = slab * SLABW + ct * 16 + n2;
                if (t < ntok) out[(size_t)t * DIM + d] = acc[r];
            }
        }
        __syncthreads();   // Bst is re-staged next slab
    }
}

extern "C" void kernel_launch(void* const* d_in, const int* in_sizes, int n_in,
                              void* d_out, int out_size, void* d_ws, size_t ws_size,
                              hipStream_t stream) {
    const float* x   = (const float*)d_in[0];
    const float* w1  = (const float*)d_in[1];
    const float* b1  = (const float*)d_in[2];
    const float* w2  = (const float*)d_in[3];
    const float* b2  = (const float*)d_in[4];
    const float* emb = (const float*)d_in[5];
    float* out = (float*)d_out;

    const int ntok = in_sizes[0];                 // B*S*1 = 65536 tokens
    const int nblocks = (ntok + TM - 1) / TM;     // 2048 blocks of 8 wave32s
    auto_disc_emb_kernel<<<nblocks, NTHREADS, 0, stream>>>(x, w1, b1, w2, b2, emb, out, ntok);
}
